// LocalPoolPointnet_21715354649482
// MI455X (gfx1250) — compile-verified, matrix operand order checked by reading, not run on hardware
//
#include <hip/hip_runtime.h>
#include <hip/hip_bf16.h>

typedef __attribute__((ext_vector_type(16))) _Float16 v16h;
typedef __attribute__((ext_vector_type(8)))  _Float16 v8h;
typedef __attribute__((ext_vector_type(8)))  float    v8f;

#define RESO_   128
#define R2_     16384
#define NBATCH_ 4
#define TPTS_   32768
#define NPTS_   (NBATCH_ * TPTS_)      // 131072
#define NBINS_  (NBATCH_ * R2_)        // 65536
#define HDIM_   128
#define CDIM_   64

// ---------------------------------------------------------------------------
// WMMA fragment loaders.
//
// 16-bit A-matrix 16x32 layout (ISA 7.12.2): lane l holds row m = l&15;
// half = l>>4 selects K-phase. In the v16h register, elements 0..7 are
// K = kbase + half*8 + e (contiguous), elements 8..15 are
// K = kbase + 16 + half*8 + (e-8) (contiguous). So a fragment is two
// contiguous 8-halfword runs -> two 16B vector loads. ReLU is pre-applied by
// the producers (Xrelu buffer), so no VALU work here.
// B operands are pre-packed so each lane's 16 values are contiguous (32B).
// ---------------------------------------------------------------------------
__device__ __forceinline__ v16h load_frag_a(const _Float16* __restrict__ rowPtr,
                                            int kbase, int half) {
  const _Float16* p0 = rowPtr + kbase + half * 8;
  v8h lo = *(const v8h*)(p0);
  v8h hi = *(const v8h*)(p0 + 16);
  v16h r;
#pragma unroll
  for (int i = 0; i < 8; ++i) { r[i] = lo[i]; r[i + 8] = hi[i]; }
  return r;
}

__device__ __forceinline__ v16h load_frag_b(const _Float16* __restrict__ Wp,
                                            int fragIdx, int lane) {
  return *(const v16h*)(Wp + ((size_t)fragIdx * 32 + lane) * 16);
}

// ---------------------------------------------------------------------------
// Weight repack: f32 [K, Ncols] row-major -> f16 fragment order.
// packed[((cb*nKB + kb)*32 + lane)*16 + e] = W[k, cb*16 + (lane&15)]
// with k = kb*32 + (e>>3)*16 + (lane>>4)*8 + (e&7).
// ---------------------------------------------------------------------------
__global__ void k_pack(const float* __restrict__ W, _Float16* __restrict__ out,
                       int K, int Ncols) {
  int tid = blockIdx.x * blockDim.x + threadIdx.x;
  if (tid >= K * Ncols) return;
  int nKB  = K >> 5;
  int frag = tid >> 9;
  int lane = (tid >> 4) & 31;
  int e    = tid & 15;
  int cb = frag / nKB, kb = frag % nKB;
  int half = lane >> 4, ln = lane & 15;
  int col = cb * 16 + ln;
  int k   = kb * 32 + ((e >> 3) << 4) + half * 8 + (e & 7);
  out[tid] = (_Float16)W[(size_t)k * Ncols + col];
}

// ---------------------------------------------------------------------------
// Plane bin indices for xz, xy, yz.
// ---------------------------------------------------------------------------
__global__ void k_indices(const float* __restrict__ p, int* __restrict__ idx) {
  int n = blockIdx.x * blockDim.x + threadIdx.x;
  if (n >= NPTS_) return;
  float x = p[n * 3 + 0], y = p[n * 3 + 1], z = p[n * 3 + 2];
  int boff = (n / TPTS_) * R2_;
  const float inv = 1.0f / (1.0f + 0.1f + 1e-5f);
  auto binOf = [&](float a, float b) {
    float ua = fminf(fmaxf(a * inv + 0.5f, 0.f), 1.f - 1e-5f);
    float ub = fminf(fmaxf(b * inv + 0.5f, 0.f), 1.f - 1e-5f);
    return (int)(ua * RESO_) + RESO_ * (int)(ub * RESO_) + boff;
  };
  idx[0 * NPTS_ + n] = binOf(x, z);
  idx[1 * NPTS_ + n] = binOf(x, y);
  idx[2 * NPTS_ + n] = binOf(y, z);
}

// ---------------------------------------------------------------------------
// fc_pos: v = b[c] + sum_d p[n,d] * W[d,c]; write raw + relu f16 copies.
// ---------------------------------------------------------------------------
__global__ void k_fcpos(const float* __restrict__ p, const float* __restrict__ W,
                        const float* __restrict__ b,
                        _Float16* __restrict__ Xraw,
                        _Float16* __restrict__ Xrelu) {
  int gid = blockIdx.x * blockDim.x + threadIdx.x;  // N*256 threads exactly
  int n = gid >> 8, c = gid & 255;
  float v = b[c] + p[n * 3 + 0] * W[c] + p[n * 3 + 1] * W[256 + c] +
            p[n * 3 + 2] * W[512 + c];
  Xraw[gid]  = (_Float16)v;
  Xrelu[gid] = (_Float16)__builtin_fmaxf(v, 0.f);
}

// ---------------------------------------------------------------------------
// Fused ResNet block, WMMA f16 -> f32.
//   T   = relu(X) @ W0 + b0          (A from Xrelu; relu(T) staged in LDS)
//   out = relu(T) @ W1 + b1 + X @ Ws (A from LDS / Xraw)
// 256 threads = 8 waves; wave w owns output cols [16w, 16w+16).
// ---------------------------------------------------------------------------
__global__ __launch_bounds__(256)
void k_resblock(const _Float16* __restrict__ Xraw,   // [N,256] f16
                const _Float16* __restrict__ Xrelu,  // [N,256] f16, relu'd
                const _Float16* __restrict__ W0p, const _Float16* __restrict__ W1p,
                const _Float16* __restrict__ WSp,
                const float* __restrict__ b0, const float* __restrict__ b1,
                float* __restrict__ netOut,          // [N,128] f32
                _Float16* __restrict__ XrawLow,      // cols 0..127
                _Float16* __restrict__ XreluLow) {   // cols 0..127
  __shared__ _Float16 lt[16 * 136];  // padded stride 136 -> conflict-free b128
  int tid = threadIdx.x;
  int wave = tid >> 5, lane = tid & 31;
  int half = lane >> 4, ln = lane & 15;
  int rowBase = blockIdx.x * 16;
  const _Float16* arowRelu = Xrelu + (size_t)(rowBase + ln) * 256;
  const _Float16* arowRaw  = Xraw  + (size_t)(rowBase + ln) * 256;

  // ---- stage 1: T = relu(X) @ W0 + b0
  v8f acc = {};
#pragma unroll
  for (int kb = 0; kb < 8; ++kb) {
    v16h a = load_frag_a(arowRelu, kb * 32, half);
    v16h b = load_frag_b(W0p, wave * 8 + kb, lane);
    acc = __builtin_amdgcn_wmma_f32_16x16x32_f16(false, a, false, b,
                                                 (short)0, acc, false, false);
  }
  float bias0 = b0[wave * 16 + ln];
#pragma unroll
  for (int r = 0; r < 8; ++r) {
    float t = __builtin_fmaxf(acc[r] + bias0, 0.f);  // relu(T), once
    lt[(r + half * 8) * 136 + wave * 16 + ln] = (_Float16)t;
  }
  __syncthreads();

  // ---- stage 2: out = relu(T) @ W1 + b1 + X @ Ws
  v8f acc2 = {};
  const _Float16* trow = &lt[ln * 136];
#pragma unroll
  for (int kb = 0; kb < 4; ++kb) {
    v16h a = load_frag_a(trow, kb * 32, half);
    v16h b = load_frag_b(W1p, wave * 4 + kb, lane);
    acc2 = __builtin_amdgcn_wmma_f32_16x16x32_f16(false, a, false, b,
                                                  (short)0, acc2, false, false);
  }
#pragma unroll
  for (int kb = 0; kb < 8; ++kb) {
    v16h a = load_frag_a(arowRaw, kb * 32, half);
    v16h b = load_frag_b(WSp, wave * 8 + kb, lane);
    acc2 = __builtin_amdgcn_wmma_f32_16x16x32_f16(false, a, false, b,
                                                  (short)0, acc2, false, false);
  }
  float bias1 = b1[wave * 16 + ln];
  int col = wave * 16 + ln;
#pragma unroll
  for (int r = 0; r < 8; ++r) {
    int row = rowBase + r + half * 8;  // D layout: M = r + (lane>>4)*8
    float v = acc2[r] + bias1;
    netOut[(size_t)row * HDIM_ + col]   = v;
    XrawLow[(size_t)row * 256 + col]    = (_Float16)v;
    XreluLow[(size_t)row * 256 + col]   = (_Float16)__builtin_fmaxf(v, 0.f);
  }
}

// ---------------------------------------------------------------------------
// fc_c: c = net @ fc_c_w + fc_c_b  ([N,128] x [128,64]); 4 waves per block.
// Reads Xraw low half (no relu before fc_c in the reference).
// ---------------------------------------------------------------------------
__global__ __launch_bounds__(128)
void k_fcc(const _Float16* __restrict__ Xraw,
           const _Float16* __restrict__ Wp, const float* __restrict__ bias,
           float* __restrict__ cOut) {      // [N,64]
  int tid = threadIdx.x;
  int wave = tid >> 5, lane = tid & 31;
  int half = lane >> 4, ln = lane & 15;
  int rowBase = blockIdx.x * 16;
  const _Float16* arow = Xraw + (size_t)(rowBase + ln) * 256;
  v8f acc = {};
#pragma unroll
  for (int kb = 0; kb < 4; ++kb) {
    v16h a = load_frag_a(arow, kb * 32, half);
    v16h b = load_frag_b(Wp, wave * 4 + kb, lane);
    acc = __builtin_amdgcn_wmma_f32_16x16x32_f16(false, a, false, b,
                                                 (short)0, acc, false, false);
  }
  float bb = bias[wave * 16 + ln];
  int col = wave * 16 + ln;
#pragma unroll
  for (int r = 0; r < 8; ++r)
    cOut[(size_t)(rowBase + r + half * 8) * CDIM_ + col] = acc[r] + bb;
}

// ---------------------------------------------------------------------------
// Pooling pipeline
// ---------------------------------------------------------------------------
__global__ void k_fill_u32(unsigned int* __restrict__ p, unsigned int v, int n) {
  int gid = blockIdx.x * blockDim.x + threadIdx.x;
  if (gid < n) p[gid] = v;
}

__device__ __forceinline__ void atomic_max_f32(float* addr, float v) {
  // classic signed/unsigned split: works for any mix of signs, -inf init
  if (v >= 0.f) atomicMax((int*)addr, __float_as_int(v));
  else          atomicMin((unsigned int*)addr, __float_as_uint(v));
}

__global__ void k_scatter_max(const float* __restrict__ net,
                              const int* __restrict__ idx,
                              float* __restrict__ bins) {  // [3][NBINS][128]
  int gid = blockIdx.x * blockDim.x + threadIdx.x;  // N*128 threads exactly
  int n = gid >> 7, h = gid & 127;
  float v = net[gid];
#pragma unroll
  for (int pl = 0; pl < 3; ++pl) {
    int b = idx[pl * NPTS_ + n];
    atomic_max_f32(bins + ((size_t)pl * NBINS_ + b) * HDIM_ + h, v);
  }
}

__global__ void k_gather(const int* __restrict__ idx,
                         const float* __restrict__ bins,
                         _Float16* __restrict__ Xraw,
                         _Float16* __restrict__ Xrelu) {  // write cols 128..255
  int gid = blockIdx.x * blockDim.x + threadIdx.x;  // N*128 threads exactly
  int n = gid >> 7, h = gid & 127;
  float s = 0.f;
#pragma unroll
  for (int pl = 0; pl < 3; ++pl) {
    int b = idx[pl * NPTS_ + n];
    s += bins[((size_t)pl * NBINS_ + b) * HDIM_ + h];
  }
  size_t o = (size_t)n * 256 + 128 + h;
  Xraw[o]  = (_Float16)s;
  Xrelu[o] = (_Float16)__builtin_fmaxf(s, 0.f);
}

__global__ void k_scatter_sum(const float* __restrict__ c,
                              const int* __restrict__ idx,
                              float* __restrict__ sums) {  // [3][NBINS][64]
  int gid = blockIdx.x * blockDim.x + threadIdx.x;  // N*64 threads exactly
  int n = gid >> 6, ch = gid & 63;
  float v = c[gid];
#pragma unroll
  for (int pl = 0; pl < 3; ++pl) {
    int b = idx[pl * NPTS_ + n];
    atomicAdd(sums + ((size_t)pl * NBINS_ + b) * CDIM_ + ch, v);
  }
}

__global__ void k_scatter_cnt(const int* __restrict__ idx, float* __restrict__ cnt) {
  int n = blockIdx.x * blockDim.x + threadIdx.x;
  if (n >= NPTS_) return;
#pragma unroll
  for (int pl = 0; pl < 3; ++pl)
    atomicAdd(cnt + pl * NBINS_ + idx[pl * NPTS_ + n], 1.0f);
}

__global__ void k_finalize(const float* __restrict__ sums,
                           const float* __restrict__ cnt,
                           float* __restrict__ out) {
  int gid = blockIdx.x * blockDim.x + threadIdx.x;  // 3*NBINS*64 = out_size
  int ch = gid & 63;
  int t  = gid >> 6;
  int g  = t % NBINS_;
  int pl = t / NBINS_;
  float cv = cnt[pl * NBINS_ + g];
  float m = sums[((size_t)pl * NBINS_ + g) * CDIM_ + ch] / fmaxf(cv, 1.0f);
  int b = g / R2_, r = g % R2_;
  out[(size_t)pl * NBATCH_ * CDIM_ * R2_ + (size_t)b * CDIM_ * R2_ +
      (size_t)ch * R2_ + r] = m;
}

// ---------------------------------------------------------------------------
// Host orchestration
// ---------------------------------------------------------------------------
extern "C" void kernel_launch(void* const* d_in, const int* in_sizes, int n_in,
                              void* d_out, int out_size, void* d_ws, size_t ws_size,
                              hipStream_t stream) {
  const float* p        = (const float*)d_in[0];
  const float* fc_pos_w = (const float*)d_in[1];
  const float* fc_pos_b = (const float*)d_in[2];
  const float* bw0      = (const float*)d_in[3];  // [5,256,128]
  const float* bb0      = (const float*)d_in[4];  // [5,128]
  const float* bw1      = (const float*)d_in[5];  // [5,128,128]
  const float* bb1      = (const float*)d_in[6];  // [5,128]
  const float* bws      = (const float*)d_in[7];  // [5,256,128]
  const float* fc_c_w   = (const float*)d_in[8];  // [128,64]
  const float* fc_c_b   = (const float*)d_in[9];  // [64]

  // workspace carve (256B aligned)
  char* base = (char*)d_ws;
  size_t off = 0;
  auto carve = [&](size_t bytes) -> void* {
    void* q = base + off;
    off = (off + bytes + 255) & ~(size_t)255;
    return q;
  };
  int*       idx    = (int*)      carve((size_t)3 * NPTS_ * 4);
  _Float16*  Xraw   = (_Float16*) carve((size_t)NPTS_ * 256 * 2);
  _Float16*  Xrelu  = (_Float16*) carve((size_t)NPTS_ * 256 * 2);
  float*     net    = (float*)    carve((size_t)NPTS_ * HDIM_ * 4);
  float*     cbuf   = (float*)    carve((size_t)NPTS_ * CDIM_ * 4);
  float*     bins   = (float*)    carve((size_t)3 * NBINS_ * HDIM_ * 4); // reused as sums
  float*     cnt    = (float*)    carve((size_t)3 * NBINS_ * 4);
  _Float16*  packed = (_Float16*) carve((size_t)417792 * 2);
  (void)ws_size; (void)n_in; (void)in_sizes; (void)out_size;

  // per-block packed weight slices: [w0(32768) | ws(32768) | w1(16384)] x5, fc_c
  auto pw0 = [&](int i) { return packed + (size_t)i * 81920; };
  auto pws = [&](int i) { return packed + (size_t)i * 81920 + 32768; };
  auto pw1 = [&](int i) { return packed + (size_t)i * 81920 + 65536; };
  _Float16* pfcc = packed + 409600;

  // 1) bin indices
  k_indices<<<(NPTS_ + 255) / 256, 256, 0, stream>>>(p, idx);

  // 2) repack all weights into WMMA fragment order (one-time per launch)
  for (int i = 0; i < 5; ++i) {
    k_pack<<<(256 * 128) / 256, 256, 0, stream>>>(bw0 + (size_t)i * 256 * 128, pw0(i), 256, 128);
    k_pack<<<(256 * 128) / 256, 256, 0, stream>>>(bws + (size_t)i * 256 * 128, pws(i), 256, 128);
    k_pack<<<(128 * 128) / 256, 256, 0, stream>>>(bw1 + (size_t)i * 128 * 128, pw1(i), 128, 128);
  }
  k_pack<<<(128 * 64) / 256, 256, 0, stream>>>(fc_c_w, pfcc, 128, 64);

  // 3) fc_pos -> Xraw/Xrelu[:, 0:256]
  k_fcpos<<<NPTS_, 256, 0, stream>>>(p, fc_pos_w, fc_pos_b, Xraw, Xrelu);

  // 4) block 0 (input = fc_pos output)
  k_resblock<<<NPTS_ / 16, 256, 0, stream>>>(Xraw, Xrelu, pw0(0), pw1(0), pws(0),
                                             bb0, bb1, net, Xraw, Xrelu);

  // 5) blocks 1..4 with scatter-max pooling
  for (int i = 1; i < 5; ++i) {
    k_fill_u32<<<(3 * NBINS_ * HDIM_) / 256, 256, 0, stream>>>(
        (unsigned int*)bins, 0xFF800000u, 3 * NBINS_ * HDIM_);  // -inf
    k_scatter_max<<<(NPTS_ * HDIM_) / 256, 256, 0, stream>>>(net, idx, bins);
    k_gather<<<(NPTS_ * HDIM_) / 256, 256, 0, stream>>>(idx, bins, Xraw, Xrelu);
    k_resblock<<<NPTS_ / 16, 256, 0, stream>>>(Xraw, Xrelu, pw0(i), pw1(i), pws(i),
                                               bb0 + i * 128, bb1 + i * 128,
                                               net, Xraw, Xrelu);
  }

  // 6) fc_c
  k_fcc<<<NPTS_ / 16, 128, 0, stream>>>(Xraw, pfcc, fc_c_b, cbuf);

  // 7) scatter-mean into three planes (sums reuse bins region)
  hipMemsetAsync(bins, 0, (size_t)3 * NBINS_ * CDIM_ * 4, stream);
  hipMemsetAsync(cnt, 0, (size_t)3 * NBINS_ * 4, stream);
  k_scatter_sum<<<(NPTS_ * CDIM_) / 256, 256, 0, stream>>>(cbuf, idx, bins);
  k_scatter_cnt<<<(NPTS_ + 255) / 256, 256, 0, stream>>>(idx, cnt);
  k_finalize<<<(3 * NBINS_ * CDIM_) / 256, 256, 0, stream>>>(bins, cnt,
                                                             (float*)d_out);
}
